// PlainMultiheadAttentionAdapter_28355374088386
// MI455X (gfx1250) — matrix operand order repaired and checked by reading, not compiled, not run on hardware
//
#include <hip/hip_runtime.h>

typedef __attribute__((ext_vector_type(16))) _Float16 v16h;
typedef __attribute__((ext_vector_type(8)))  _Float16 v8h;
typedef __attribute__((ext_vector_type(4)))  _Float16 v4h;
typedef __attribute__((ext_vector_type(8)))  float    v8f;
typedef __attribute__((ext_vector_type(4)))  unsigned int u32x4;
typedef __attribute__((ext_vector_type(8)))  int          i32x8;
typedef __attribute__((ext_vector_type(4)))  int          i32x4;

union Frag16 {
  v16h v;
  v8h  h[2];
  _Float16 e[16];
};

#define B_ 2
#define S_ 2048
#define E_ 1024
#define H_ 16
#define D_ 64
#define M_ (B_ * S_)   // 4096 rows

// Tensor Data Mover availability / arity (probe-verified discriminator:
// the TDM header ships only with the 6-arg amdgpu-toolchain).
#if defined(__HIP_DEVICE_COMPILE__) && __has_builtin(__builtin_amdgcn_tensor_load_to_lds)
#define HAVE_TDM 1
#if __has_include(<hip/amd_detail/amd_gfx1250_TDM.h>)
#define TDM_ARITY6 1
#endif
#endif

__device__ __forceinline__ v4h load4h(const float* p) {
  const float4 f = *(const float4*)p;
  v4h r;
  r.x = (_Float16)f.x; r.y = (_Float16)f.y;
  r.z = (_Float16)f.z; r.w = (_Float16)f.w;
  return r;
}
__device__ __forceinline__ v4h load4h(const _Float16* p) {
  return *(const v4h*)p;
}

// ---------------------------------------------------------------------------
// Generic Linear: Y = X @ W^T + bias.
//   MODE 0: store f16 head-split [B,H,S,D] (Q/K/V caches)
//   MODE 1: store f32 row-major [M_, E_] (output projection)
// Block = 128 threads (4 waves), block tile 64x64, K-step 32.
// ---------------------------------------------------------------------------
template <typename TIn, int MODE>
__global__ __launch_bounds__(128) void linear64x64(
    const TIn* __restrict__ X, const float* __restrict__ W,
    const float* __restrict__ bias, void* __restrict__ Yout) {
  __shared__ _Float16 As[64][40];  // pitch 40 halves = 80B
  __shared__ _Float16 Bs[64][40];

  const int tid  = threadIdx.x;
  const int wid  = tid >> 5;
  const int lane = tid & 31;
  const int l16  = lane & 15;
  const int half = lane >> 4;
  const int n0 = blockIdx.x * 64;
  const int m0 = blockIdx.y * 64;

  v8f acc[4] = {};

  for (int k0 = 0; k0 < E_; k0 += 32) {
    __syncthreads();
    // Stage A (X rows) and B (W rows: B[k][n] = W[n][k]) as f16, 64x32 each,
    // 4-element vector chunks (float4 load -> 8B LDS store).
#pragma unroll
    for (int i = 0; i < 4; ++i) {
      int idx4 = tid + i * 128;      // 0..511 quads (512 * 4 = 2048 elements)
      int r = idx4 >> 3;             // 8 quads per 32-wide row
      int c = (idx4 & 7) * 4;
      *(v4h*)&As[r][c] = load4h(&X[(size_t)(m0 + r) * E_ + k0 + c]);
      *(v4h*)&Bs[r][c] = load4h(&W[(size_t)(n0 + r) * E_ + k0 + c]);
    }
    __syncthreads();

    // A fragment (16x32): lane row = l16, k = {8h..8h+7} and {16+8h..16+8h+7}
    Frag16 a;
    const int arow = 16 * wid + l16;
    a.h[0] = *(const v8h*)&As[arow][8 * half];
    a.h[1] = *(const v8h*)&As[arow][16 + 8 * half];

#pragma unroll
    for (int nt = 0; nt < 4; ++nt) {
      Frag16 b;
      const int brow = 16 * nt + l16;
      b.h[0] = *(const v8h*)&Bs[brow][8 * half];
      b.h[1] = *(const v8h*)&Bs[brow][16 + 8 * half];
      acc[nt] = __builtin_amdgcn_wmma_f32_16x16x32_f16(
          false, a.v, false, b.v, (short)0, acc[nt], false, false);
    }
  }

#pragma unroll
  for (int nt = 0; nt < 4; ++nt) {
    const int col = n0 + 16 * nt + l16;
    const float bval = bias[col];
#pragma unroll
    for (int j = 0; j < 8; ++j) {
      const int row = m0 + 16 * wid + 8 * half + j;
      const float val = acc[nt][j] + bval;
      if (MODE == 0) {
        const int bb = row / S_, ss = row % S_;
        const int hh = col >> 6, dd = col & 63;
        ((_Float16*)Yout)[(((size_t)bb * H_ + hh) * S_ + ss) * D_ + dd] =
            (_Float16)val;
      } else {
        ((float*)Yout)[(size_t)row * E_ + col] = val;
      }
    }
  }
}

// ---------------------------------------------------------------------------
// Flash attention: block (4 waves) owns 64 query rows of one (b,h) head.
// Each 32-key tile: K staged in LDS via Tensor Data Mover (row-major 32x64),
// V staged transposed Vt[d][key] by cooperative copy. Online softmax with
// WMMA-based row sums (P @ ones).
// Q/K/V caches f16 [B,H,S,D]; output f16 [B,S,H*D] for the final projection.
// ---------------------------------------------------------------------------
__global__ __launch_bounds__(128) void flash_attn(
    const _Float16* __restrict__ Q, const _Float16* __restrict__ Kc,
    const _Float16* __restrict__ Vc, _Float16* __restrict__ O) {
  __shared__ _Float16 Ks[32][64];       // K tile, row-major (TDM destination)
  __shared__ _Float16 Vt[64][40];       // V tile, transposed [d][key]
  __shared__ _Float16 Plds[4][16][40];  // per-wave P tile (16x32)

  const int tid  = threadIdx.x;
  const int wid  = tid >> 5;
  const int lane = tid & 31;
  const int l16  = lane & 15;
  const int half = lane >> 4;

  const int bh = blockIdx.x >> 5;       // (b*H + h); 32 q-blocks per head
  const int q0 = (blockIdx.x & 31) * 64;

  const _Float16* Qh = Q  + (size_t)bh * S_ * D_;
  const _Float16* Kh = Kc + (size_t)bh * S_ * D_;
  const _Float16* Vh = Vc + (size_t)bh * S_ * D_;

  // Q A-fragments for the two 32-wide d-windows (per-wave 16-row strip).
  Frag16 aq[2];
  {
    const _Float16* qr = Qh + (size_t)(q0 + 16 * wid + l16) * D_;
#pragma unroll
    for (int f = 0; f < 2; ++f) {
      aq[f].h[0] = *(const v8h*)(qr + 32 * f + 8 * half);
      aq[f].h[1] = *(const v8h*)(qr + 32 * f + 16 + 8 * half);
    }
  }

  // all-ones B fragment for WMMA row sums: T = P @ ones -> T[m][*] = rowsum
  Frag16 ones;
#pragma unroll
  for (int i = 0; i < 16; ++i) ones.e[i] = (_Float16)1.0f;

  float rm[8], rs[8];
  v8f o[4] = {};
#pragma unroll
  for (int j = 0; j < 8; ++j) { rm[j] = -3.0e38f; rs[j] = 0.0f; }

  const float scale = 0.125f;  // 1/sqrt(64)
  const int skey = tid >> 2;          // staging: key row 0..31
  const int sd0  = (tid & 3) * 16;    // staging: d chunk base

  for (int kb0 = 0; kb0 < S_; kb0 += 32) {
    __syncthreads();  // previous tile fully consumed

    if (kb0 + 32 < S_) {  // speculative prefetch of next tiles
      __builtin_prefetch(Kh + (size_t)(kb0 + 32) * D_, 0, 0);
      __builtin_prefetch(Vh + (size_t)(kb0 + 32) * D_, 0, 0);
    }

    // ---- stage V transposed: Vt[d][key] (coalesced v8h loads) ----
    {
      const _Float16* vr = Vh + (size_t)(kb0 + skey) * D_ + sd0;
      v8h va = *(const v8h*)(vr);
      v8h vb = *(const v8h*)(vr + 8);
#pragma unroll
      for (int i = 0; i < 8; ++i) {
        Vt[sd0 + i][skey]     = va[i];
        Vt[sd0 + 8 + i][skey] = vb[i];
      }
    }

    // ---- stage K tile (row-major 32x64) ----
#if HAVE_TDM
    if (wid == 0) {
      // Tensor DMA descriptor (CDNA5 ISA):
      // 2D tensor [S_, 64] f16, tile 32 rows x 64 cols at row kb0.
      const uint32_t ldsa = (uint32_t)(uintptr_t)(&Ks[0][0]);
      const uint64_t ga   = (uint64_t)(uintptr_t)(Kh + (size_t)kb0 * D_);
      u32x4 g0;
      g0[0] = 1u;                                   // count=1, user mode
      g0[1] = ldsa;                                 // lds_addr
      g0[2] = (uint32_t)ga;                         // global_addr[31:0]
      g0[3] = (uint32_t)((ga >> 32) & 0x1FFFFFFu)   // global_addr[56:32]
              | (2u << 30);                         // type=2 ("image")
      i32x8 g1;
      g1[0] = (int)(1u << 16);                      // data_size=1 -> 2 bytes
      g1[1] = (int)(64u << 16);                     // tensor_dim0 = 64
      g1[2] = (int)(((unsigned)S_ & 0xFFFFu) << 16);// tensor_dim1 = S_
      g1[3] = (int)(64u << 16);                     // tile_dim0 = 64
      g1[4] = 32;                                   // tile_dim1 = 32
      g1[5] = 64;                                   // tensor_dim0_stride = 64
      g1[6] = 0;
      g1[7] = 0;
      i32x4 gz4 = {0, 0, 0, 0};
#if TDM_ARITY6
      i32x8 gz8 = {0, 0, 0, 0, 0, 0, 0, 0};
      __builtin_amdgcn_tensor_load_to_lds(g0, g1, gz4, gz4, gz8, 0);
#else
      __builtin_amdgcn_tensor_load_to_lds(g0, g1, gz4, gz4, 0);
#endif
      __builtin_amdgcn_s_wait_tensorcnt(0);
    }
#else
    {
      const _Float16* kr = Kh + (size_t)(kb0 + skey) * D_ + sd0;
      *(v8h*)&Ks[skey][sd0]     = *(const v8h*)(kr);
      *(v8h*)&Ks[skey][sd0 + 8] = *(const v8h*)(kr + 8);
    }
#endif
    __syncthreads();

    // ---- scores: Q(16x64) @ K^T -> two 16x16 N-tiles over 32 keys ----
    v8f c[2];
#pragma unroll
    for (int nt = 0; nt < 2; ++nt) {
      const int key = 16 * nt + l16;
      v8f cc = {};
#pragma unroll
      for (int ks = 0; ks < 2; ++ks) {
        Frag16 b;
        b.h[0] = *(const v8h*)&Ks[key][32 * ks + 8 * half];
        b.h[1] = *(const v8h*)&Ks[key][32 * ks + 16 + 8 * half];
        cc = __builtin_amdgcn_wmma_f32_16x16x32_f16(
            false, aq[ks].v, false, b.v, (short)0, cc, false, false);
      }
      c[nt] = cc;
    }

    // ---- online softmax max pass (rows m = 8*half + j in this half-group);
    //      row sums are done on the matrix pipe below ----
    float alpha[8];
#pragma unroll
    for (int j = 0; j < 8; ++j) {
      float s0 = c[0][j] * scale;
      float s1 = c[1][j] * scale;
      float mx = fmaxf(s0, s1);
      mx = fmaxf(mx, __shfl_xor(mx, 1, 32));
      mx = fmaxf(mx, __shfl_xor(mx, 2, 32));
      mx = fmaxf(mx, __shfl_xor(mx, 4, 32));
      mx = fmaxf(mx, __shfl_xor(mx, 8, 32));
      const float mnew = fmaxf(rm[j], mx);
      alpha[j] = __expf(rm[j] - mnew);
      rm[j] = mnew;
      const float p0 = __expf(s0 - mnew);
      const float p1 = __expf(s1 - mnew);
#pragma unroll
      for (int dt = 0; dt < 4; ++dt) o[dt][j] *= alpha[j];
      Plds[wid][8 * half + j][l16]      = (_Float16)p0;
      Plds[wid][8 * half + j][l16 + 16] = (_Float16)p1;
    }

    // ---- reshape P (C-layout) -> A-fragment via per-wave LDS tile ----
    Frag16 pa;
    pa.h[0] = *(const v8h*)&Plds[wid][l16][8 * half];
    pa.h[1] = *(const v8h*)&Plds[wid][l16][16 + 8 * half];

    // ---- row sums on the matrix pipe: T = P @ ones, T[j] = rowsum(m) ----
    {
      v8f t = {};
      t = __builtin_amdgcn_wmma_f32_16x16x32_f16(
          false, pa.v, false, ones.v, (short)0, t, false, false);
#pragma unroll
      for (int j = 0; j < 8; ++j) rs[j] = rs[j] * alpha[j] + t[j];
    }

    // ---- P(16x32) @ V(32x64): B-fragments from transposed Vt ----
#pragma unroll
    for (int dt = 0; dt < 4; ++dt) {
      Frag16 bfr;
      const int d = 16 * dt + l16;
      bfr.h[0] = *(const v8h*)&Vt[d][8 * half];
      bfr.h[1] = *(const v8h*)&Vt[d][16 + 8 * half];
      o[dt] = __builtin_amdgcn_wmma_f32_16x16x32_f16(
          false, pa.v, false, bfr.v, (short)0, o[dt], false, false);
    }
  }

  // ---- epilogue: divide by denominator, store as [B,S,H,D] ----
  const int bb = bh >> 4;
  const int hh = bh & 15;
#pragma unroll
  for (int dt = 0; dt < 4; ++dt) {
#pragma unroll
    for (int j = 0; j < 8; ++j) {
      const int srow = q0 + 16 * wid + 8 * half + j;
      const float val = o[dt][j] / rs[j];
      O[(((size_t)bb * S_ + srow) * H_ + hh) * D_ + 16 * dt + l16] =
          (_Float16)val;
    }
  }
}

// ---------------------------------------------------------------------------
extern "C" void kernel_launch(void* const* d_in, const int* in_sizes, int n_in,
                              void* d_out, int out_size, void* d_ws,
                              size_t ws_size, hipStream_t stream) {
  (void)in_sizes; (void)n_in; (void)out_size; (void)ws_size;
  const float* query = (const float*)d_in[0];
  const float* key_  = (const float*)d_in[1];
  const float* value = (const float*)d_in[2];
  const float* Wq = (const float*)d_in[3];
  const float* bq = (const float*)d_in[4];
  const float* Wk = (const float*)d_in[5];
  const float* bk = (const float*)d_in[6];
  const float* Wv = (const float*)d_in[7];
  const float* bv = (const float*)d_in[8];
  const float* Wo = (const float*)d_in[9];
  const float* bo = (const float*)d_in[10];

  char* ws = (char*)d_ws;
  const size_t sz = (size_t)M_ * E_ * sizeof(_Float16);  // 8 MB each
  _Float16* q16 = (_Float16*)(ws + 0 * sz);
  _Float16* k16 = (_Float16*)(ws + 1 * sz);
  _Float16* v16 = (_Float16*)(ws + 2 * sz);
  _Float16* a16 = (_Float16*)(ws + 3 * sz);

  dim3 gridL(E_ / 64, M_ / 64);  // (16, 64)
  dim3 blk(128);

  linear64x64<float, 0><<<gridL, blk, 0, stream>>>(query, Wq, bq, q16);
  linear64x64<float, 0><<<gridL, blk, 0, stream>>>(key_,  Wk, bk, k16);
  linear64x64<float, 0><<<gridL, blk, 0, stream>>>(value, Wv, bv, v16);

  flash_attn<<<dim3(B_ * H_ * (S_ / 64)), blk, 0, stream>>>(q16, k16, v16,
                                                            a16);

  linear64x64<_Float16, 1><<<gridL, blk, 0, stream>>>(a16, Wo, bo, d_out);
}
